// GraphEncoder_53678501265724
// MI455X (gfx1250) — matrix operand order repaired and checked by reading, not compiled, not run on hardware
//
#include <hip/hip_runtime.h>
#include <hip/hip_bf16.h>

#define N_NODES 100000
#define N_EDGES 600000
#define DIM     128
#define NHID_   512
#define NOUT_   768
#define NGRAPH  2048
#define EPS_    1e-5f

typedef __attribute__((ext_vector_type(16))) __bf16 v16bf;
typedef __attribute__((ext_vector_type(8)))  float  v8f;

__device__ __forceinline__ float warp_sum(float v) {
  #pragma unroll
  for (int o = 16; o > 0; o >>= 1) v += __shfl_xor(v, o, 32);
  return v;
}

// ------------------------- utility kernels -------------------------
__global__ void fill_kernel(float* __restrict__ p, float v, int n) {
  int i = blockIdx.x * blockDim.x + threadIdx.x;
  if (i < n) p[i] = v;
}
__global__ void copy_kernel(float* __restrict__ d, const float* __restrict__ s, int n) {
  int i = blockIdx.x * blockDim.x + threadIdx.x;
  if (i < n) d[i] = s[i];
}
__global__ void bias_rows_kernel(float* __restrict__ y, const float* __restrict__ b,
                                 int total) {
  int i = blockIdx.x * blockDim.x + threadIdx.x;
  if (i < total) y[i] = b[i & (DIM - 1)];
}

// ------------- WMMA GEMM: C = A(MxK) @ W(KxN) [+bias][relu] -------------
// bf16 inputs (converted on load), fp32 accumulate via v_wmma_f32_16x16x32_bf16.
// K is a template constant -> K-loop fully unrolls (back-to-back WMMAs).
// Each wave computes TWO adjacent 16x16 tiles (A-fragment reused by 2 WMMAs);
// block = 4 waves = one 16x128 output strip. Requires M%16==0, K%32==0, N%128==0.
template<int K>
__global__ __launch_bounds__(128) void gemm_wmma_bf16(
    const float* __restrict__ A, const float* __restrict__ W,
    const float* __restrict__ bias, float* __restrict__ C,
    int N, int relu) {
  const int lane = threadIdx.x & 31;
  const int wave = threadIdx.x >> 5;
  const int l16  = lane & 15;
  const int h    = lane >> 4;                       // half-wave
  const int mt   = blockIdx.x << 4;                 // row tile base
  const int nt0  = (blockIdx.y * 8 + wave * 2) << 4;  // first of two col tiles

  v8f acc0 = {}, acc1 = {};
  const float* arow = A + (size_t)(mt + l16) * K + h * 16;
  #pragma unroll
  for (int k0 = 0; k0 < K; k0 += 32) {
    v16bf afrag, b0, b1;
    // A 16x32 bf16 fragment: lane half h holds K = k0 + 16h + j
    #pragma unroll
    for (int j = 0; j < 8; ++j) afrag[j]     = (__bf16)arow[k0 + j];
    #pragma unroll
    for (int j = 0; j < 8; ++j) afrag[8 + j] = (__bf16)arow[k0 + 8 + j];
    // B 32x16 bf16 fragments: col = nt0(+16)+l16, K = k0 + 16h + j (W row-major)
    const float* wp = W + (size_t)(k0 + h * 16) * N + (nt0 + l16);
    #pragma unroll
    for (int j = 0; j < 16; ++j) b0[j] = (__bf16)wp[(size_t)j * N];
    #pragma unroll
    for (int j = 0; j < 16; ++j) b1[j] = (__bf16)wp[(size_t)j * N + 16];

    acc0 = __builtin_amdgcn_wmma_f32_16x16x32_bf16(false, afrag, false, b0,
                                                   (short)0, acc0, false, false);
    acc1 = __builtin_amdgcn_wmma_f32_16x16x32_bf16(false, afrag, false, b1,
                                                   (short)0, acc1, false, false);
  }
  const float bv0 = bias ? bias[nt0 + l16]      : 0.0f;
  const float bv1 = bias ? bias[nt0 + 16 + l16] : 0.0f;
  #pragma unroll
  for (int r = 0; r < 8; ++r) {                     // C/D: row = r + 8h, col = l16
    float v0 = acc0[r] + bv0;
    float v1 = acc1[r] + bv1;
    if (relu) { v0 = fmaxf(v0, 0.0f); v1 = fmaxf(v1, 0.0f); }
    float* cp = C + (size_t)(mt + r + 8 * h) * N + (nt0 + l16);
    cp[0]  = v0;
    cp[16] = v1;
  }
}

// ------------------------- graph kernels -------------------------
__global__ void deg_kernel(const int* __restrict__ dst, float* __restrict__ deg, int E) {
  int i = blockIdx.x * blockDim.x + threadIdx.x;
  if (i < E) atomicAdd(&deg[dst[i]], 1.0f);
}
__global__ void rsqrt_kernel(float* __restrict__ p, int n) {
  int i = blockIdx.x * blockDim.x + threadIdx.x;
  if (i < n) p[i] = rsqrtf(p[i]);
}

// y[dst] += h[src] * dinv[src]*dinv[dst]; thread per (edge, 4-feature group)
__global__ void gcn_scatter_kernel(const float* __restrict__ h, const int* __restrict__ ei,
                                   const float* __restrict__ dinv, float* __restrict__ y,
                                   int E, int Nn) {
  int i = blockIdx.x * blockDim.x + threadIdx.x;
  int total = (E + Nn) * 32;
  if (i >= total) return;
  int e = i >> 5;
  int f = (i & 31) << 2;
  int s, d;
  if (e < E) { s = ei[e]; d = ei[E + e]; } else { s = d = e - E; }
  float nrm = dinv[s] * dinv[d];
  const float4 hv = *(const float4*)(h + (size_t)s * DIM + f);
  float* yp = y + (size_t)d * DIM + f;
  atomicAdd(yp + 0, hv.x * nrm);
  atomicAdd(yp + 1, hv.y * nrm);
  atomicAdd(yp + 2, hv.z * nrm);
  atomicAdd(yp + 3, hv.w * nrm);
}

// ------------------------- batchnorm -------------------------
__global__ __launch_bounds__(128) void bn_stats_kernel(const float* __restrict__ y,
    float* __restrict__ sum, float* __restrict__ sumsq, int rows) {
  int f = threadIdx.x;                         // 128 features, coalesced
  float s = 0.f, ss = 0.f;
  for (int r = blockIdx.x; r < rows; r += gridDim.x) {
    float v = y[(size_t)r * DIM + f];
    s += v; ss += v * v;
  }
  atomicAdd(&sum[f], s);
  atomicAdd(&sumsq[f], ss);
}
__global__ void bn_finalize_kernel(const float* __restrict__ sum, const float* __restrict__ sumsq,
                                   float* __restrict__ mean, float* __restrict__ rstd, int rows) {
  int f = threadIdx.x;
  float mu  = sum[f] / (float)rows;
  float var = sumsq[f] / (float)rows - mu * mu;
  mean[f] = mu;
  rstd[f] = rsqrtf(var + EPS_);
}
__global__ void bn_apply_kernel(const float* __restrict__ y, const float* __restrict__ mean,
                                const float* __restrict__ rstd, const float* __restrict__ g,
                                const float* __restrict__ be, const float* __restrict__ res,
                                float* __restrict__ out, int total) {
  int i = blockIdx.x * blockDim.x + threadIdx.x;
  if (i >= total) return;
  int f = i & (DIM - 1);
  float v = (y[i] - mean[f]) * rstd[f] * g[f] + be[f];
  v = fmaxf(v, 0.f);
  out[i] = res ? v + res[i] : v;
}

// ------------------------- GAT attention -------------------------
__global__ __launch_bounds__(256) void gat_dots_kernel(const float* __restrict__ h,
    const float* __restrict__ asrc, const float* __restrict__ adst,
    float* __restrict__ al, float* __restrict__ ar, int n) {
  int w = (blockIdx.x * blockDim.x + threadIdx.x) >> 5;  // one wave32 per node
  int lane = threadIdx.x & 31;
  if (w >= n) return;
  const float* row = h + (size_t)w * DIM;
  float s1 = 0.f, s2 = 0.f;
  for (int j = lane; j < DIM; j += 32) {
    float v = row[j];
    s1 += v * asrc[j];
    s2 += v * adst[j];
  }
  s1 = warp_sum(s1);
  s2 = warp_sum(s2);
  if (lane == 0) { al[w] = s1; ar[w] = s2; }
}
__device__ __forceinline__ float leaky02(float v) { return v > 0.f ? v : 0.2f * v; }
__device__ __forceinline__ void atomicMaxF(float* addr, float v) {
  if (v >= 0.f) atomicMax((int*)addr, __float_as_int(v));
  else          atomicMin((unsigned int*)addr, (unsigned int)__float_as_int(v));
}
__global__ void gat_max_kernel(const float* __restrict__ al, const float* __restrict__ ar,
                               const int* __restrict__ ei, float* __restrict__ m, int E, int Nn) {
  int e = blockIdx.x * blockDim.x + threadIdx.x;
  if (e >= E + Nn) return;
  int s, d; if (e < E) { s = ei[e]; d = ei[E + e]; } else { s = d = e - E; }
  atomicMaxF(&m[d], leaky02(al[s] + ar[d]));
}
__global__ void gat_expsum_kernel(const float* __restrict__ al, const float* __restrict__ ar,
                                  const float* __restrict__ m, const int* __restrict__ ei,
                                  float* __restrict__ denom, float* __restrict__ alpha,
                                  int E, int Nn) {
  int e = blockIdx.x * blockDim.x + threadIdx.x;
  if (e >= E + Nn) return;
  int s, d; if (e < E) { s = ei[e]; d = ei[E + e]; } else { s = d = e - E; }
  float ee = expf(leaky02(al[s] + ar[d]) - m[d]);
  alpha[e] = ee;
  atomicAdd(&denom[d], ee);
}
__global__ void gat_scatter_kernel(const float* __restrict__ h, const float* __restrict__ alpha,
                                   const float* __restrict__ denom, const int* __restrict__ ei,
                                   float* __restrict__ y, int E, int Nn) {
  int i = blockIdx.x * blockDim.x + threadIdx.x;
  int total = (E + Nn) * 32;
  if (i >= total) return;
  int e = i >> 5;
  int f = (i & 31) << 2;
  int s, d;
  if (e < E) { s = ei[e]; d = ei[E + e]; } else { s = d = e - E; }
  float w = alpha[e] / denom[d];
  const float4 hv = *(const float4*)(h + (size_t)s * DIM + f);
  float* yp = y + (size_t)d * DIM + f;
  atomicAdd(yp + 0, hv.x * w);
  atomicAdd(yp + 1, hv.y * w);
  atomicAdd(yp + 2, hv.z * w);
  atomicAdd(yp + 3, hv.w * w);
}

// ------------------------- pooling & layernorm -------------------------
__global__ void pool_kernel(const float* __restrict__ x, const int* __restrict__ batch,
                            float* __restrict__ pooled, int total) {
  int i = blockIdx.x * blockDim.x + threadIdx.x;
  if (i >= total) return;
  int n = i >> 7;               // / DIM
  int f = i & (DIM - 1);
  atomicAdd(&pooled[(size_t)batch[n] * DIM + f], x[i]);
}
__global__ __launch_bounds__(256) void layernorm_kernel(const float* __restrict__ xin,
    const float* __restrict__ g, const float* __restrict__ b,
    float* __restrict__ out, int rows, int cols) {
  int w = (blockIdx.x * blockDim.x + threadIdx.x) >> 5;  // wave32 per row
  int lane = threadIdx.x & 31;
  if (w >= rows) return;
  const float* row = xin + (size_t)w * cols;
  float s = 0.f, ss = 0.f;
  for (int j = lane; j < cols; j += 32) { float v = row[j]; s += v; ss += v * v; }
  s  = warp_sum(s);
  ss = warp_sum(ss);
  float mu   = s / (float)cols;
  float rstd = rsqrtf(ss / (float)cols - mu * mu + EPS_);
  float* orow = out + (size_t)w * cols;
  for (int j = lane; j < cols; j += 32) orow[j] = (row[j] - mu) * rstd * g[j] + b[j];
}

// ------------------------- host launcher -------------------------
extern "C" void kernel_launch(void* const* d_in, const int* in_sizes, int n_in,
                              void* d_out, int out_size, void* d_ws, size_t ws_size,
                              hipStream_t stream) {
  (void)in_sizes; (void)n_in; (void)out_size; (void)ws_size;
  const float* x     = (const float*)d_in[0];
  const int*   ei    = (const int*)d_in[1];     // [2,E] flattened: src=ei[e], dst=ei[E+e]
  const int*   batch = (const int*)d_in[2];
  const float* Wg[3] = {(const float*)d_in[3], (const float*)d_in[5], (const float*)d_in[7]};
  const float* bg[3] = {(const float*)d_in[4], (const float*)d_in[6], (const float*)d_in[8]};
  const float* Wa    = (const float*)d_in[9];
  const float* ba    = (const float*)d_in[10];
  const float* a_src = (const float*)d_in[11];
  const float* a_dst = (const float*)d_in[12];
  const float* gk[4]  = {(const float*)d_in[13], (const float*)d_in[15],
                         (const float*)d_in[17], (const float*)d_in[19]};
  const float* bek[4] = {(const float*)d_in[14], (const float*)d_in[16],
                         (const float*)d_in[18], (const float*)d_in[20]};
  const float* Wl1 = (const float*)d_in[21]; const float* bl1 = (const float*)d_in[22];
  const float* Wl2 = (const float*)d_in[23]; const float* bl2 = (const float*)d_in[24];
  const float* Wl3 = (const float*)d_in[25]; const float* bl3 = (const float*)d_in[26];
  const float* Wl4 = (const float*)d_in[27]; const float* bl4 = (const float*)d_in[28];
  const float* gln = (const float*)d_in[29]; const float* bln = (const float*)d_in[30];

  // workspace carve-up (floats)
  float* ws = (float*)d_ws;
  size_t off = 0;
  const size_t ND = (size_t)N_NODES * DIM;
  float* xcur  = ws + off; off += ND;
  float* hbuf  = ws + off; off += ND;
  float* ybuf  = ws + off; off += ND;
  float* dinv  = ws + off; off += N_NODES;
  float* al    = ws + off; off += N_NODES;
  float* ar    = ws + off; off += N_NODES;
  float* mlog  = ws + off; off += N_NODES;
  float* denom = ws + off; off += N_NODES;
  float* alpha = ws + off; off += (size_t)(N_EDGES + N_NODES);
  float* bnsum = ws + off; off += DIM;   // bnsum & bnsq contiguous (zeroed together)
  float* bnsq  = ws + off; off += DIM;
  float* bnmu  = ws + off; off += DIM;
  float* bnrs  = ws + off; off += DIM;
  float* pooled = ws + off; off += (size_t)NGRAPH * DIM;
  float* m1 = ws + off; off += (size_t)NGRAPH * NHID_;
  float* m2 = ws + off; off += (size_t)NGRAPH * NHID_;
  float* m3 = ws + off; off += (size_t)NGRAPH * NHID_;
  float* m4 = ws + off; off += (size_t)NGRAPH * NOUT_;

  const int EA = N_EDGES + N_NODES;
  const int scat_total = EA * 32;
  const int ndTot = (int)ND;
  auto blk = [](int n) { return (n + 255) / 256; };

  // x -> xcur (kept mutable for residuals)
  copy_kernel<<<blk(ndTot), 256, 0, stream>>>(xcur, x, ndTot);

  // symmetric-norm degrees (self-loop => deg starts at 1)
  fill_kernel<<<blk(N_NODES), 256, 0, stream>>>(dinv, 1.0f, N_NODES);
  deg_kernel<<<blk(N_EDGES), 256, 0, stream>>>(ei + N_EDGES, dinv, N_EDGES);
  rsqrt_kernel<<<blk(N_NODES), 256, 0, stream>>>(dinv, N_NODES);

  // ---- 3 GCN layers: WMMA GEMM -> scatter -> BN -> relu+residual ----
  for (int l = 0; l < 3; ++l) {
    gemm_wmma_bf16<DIM><<<dim3(N_NODES / 16, DIM / 128), 128, 0, stream>>>(
        xcur, Wg[l], nullptr, hbuf, DIM, 0);
    bias_rows_kernel<<<blk(ndTot), 256, 0, stream>>>(ybuf, bg[l], ndTot);
    gcn_scatter_kernel<<<blk(scat_total), 256, 0, stream>>>(hbuf, ei, dinv, ybuf,
                                                            N_EDGES, N_NODES);
    fill_kernel<<<1, 256, 0, stream>>>(bnsum, 0.f, 2 * DIM);
    bn_stats_kernel<<<512, 128, 0, stream>>>(ybuf, bnsum, bnsq, N_NODES);
    bn_finalize_kernel<<<1, DIM, 0, stream>>>(bnsum, bnsq, bnmu, bnrs, N_NODES);
    bn_apply_kernel<<<blk(ndTot), 256, 0, stream>>>(ybuf, bnmu, bnrs, gk[l], bek[l],
                                                    xcur, xcur, ndTot);
  }

  // ---- GAT layer ----
  gemm_wmma_bf16<DIM><<<dim3(N_NODES / 16, DIM / 128), 128, 0, stream>>>(
      xcur, Wa, nullptr, hbuf, DIM, 0);
  gat_dots_kernel<<<blk(N_NODES * 32), 256, 0, stream>>>(hbuf, a_src, a_dst, al, ar, N_NODES);
  fill_kernel<<<blk(N_NODES), 256, 0, stream>>>(mlog, -__builtin_inff(), N_NODES);
  fill_kernel<<<blk(N_NODES), 256, 0, stream>>>(denom, 0.f, N_NODES);
  gat_max_kernel<<<blk(EA), 256, 0, stream>>>(al, ar, ei, mlog, N_EDGES, N_NODES);
  gat_expsum_kernel<<<blk(EA), 256, 0, stream>>>(al, ar, mlog, ei, denom, alpha,
                                                 N_EDGES, N_NODES);
  bias_rows_kernel<<<blk(ndTot), 256, 0, stream>>>(ybuf, ba, ndTot);
  gat_scatter_kernel<<<blk(scat_total), 256, 0, stream>>>(hbuf, alpha, denom, ei, ybuf,
                                                          N_EDGES, N_NODES);
  fill_kernel<<<1, 256, 0, stream>>>(bnsum, 0.f, 2 * DIM);
  bn_stats_kernel<<<512, 128, 0, stream>>>(ybuf, bnsum, bnsq, N_NODES);
  bn_finalize_kernel<<<1, DIM, 0, stream>>>(bnsum, bnsq, bnmu, bnrs, N_NODES);
  bn_apply_kernel<<<blk(ndTot), 256, 0, stream>>>(ybuf, bnmu, bnrs, gk[3], bek[3],
                                                  nullptr, xcur, ndTot);

  // ---- global add pool ----
  fill_kernel<<<blk(NGRAPH * DIM), 256, 0, stream>>>(pooled, 0.f, NGRAPH * DIM);
  pool_kernel<<<blk(ndTot), 256, 0, stream>>>(xcur, batch, pooled, ndTot);

  // ---- MLP head (all WMMA) ----
  gemm_wmma_bf16<DIM><<<dim3(NGRAPH / 16, NHID_ / 128), 128, 0, stream>>>(
      pooled, Wl1, bl1, m1, NHID_, 1);
  gemm_wmma_bf16<NHID_><<<dim3(NGRAPH / 16, NHID_ / 128), 128, 0, stream>>>(
      m1, Wl2, bl2, m2, NHID_, 1);
  gemm_wmma_bf16<NHID_><<<dim3(NGRAPH / 16, NHID_ / 128), 128, 0, stream>>>(
      m2, Wl3, bl3, m3, NHID_, 1);
  gemm_wmma_bf16<NHID_><<<dim3(NGRAPH / 16, NOUT_ / 128), 128, 0, stream>>>(
      m3, Wl4, bl4, m4, NOUT_, 0);

  // ---- final layernorm over 768 ----
  layernorm_kernel<<<blk(NGRAPH * 32), 256, 0, stream>>>(m4, gln, bln, (float*)d_out,
                                                         NGRAPH, NOUT_);
}
